// AttentionShift_28518582845717
// MI455X (gfx1250) — matrix operand order repaired and checked by reading, not compiled, not run on hardware
//
#include <hip/hip_runtime.h>
#include <hip/hip_bf16.h>

// Problem constants (from reference setup_inputs)
#define B_ 32
#define P_ 32
#define N_ 4096
#define D_ 768
#define M_ 4096
#define EPS_ 1e-8f
#define TEMP_ 0.1f

typedef __attribute__((ext_vector_type(16))) __bf16 v16bf;
typedef __attribute__((ext_vector_type(8)))  __bf16 v8bf;
typedef __attribute__((ext_vector_type(8)))  float  v8f;

static __device__ __forceinline__ __bf16 f2bf(float x) {
    unsigned u = __builtin_bit_cast(unsigned, x);
    unsigned r = (u + 0x7FFFu + ((u >> 16) & 1u)) >> 16;
    unsigned short s = (unsigned short)r;
    return __builtin_bit_cast(__bf16, s);
}

// ---------------------------------------------------------------------------
// fp32 -> bf16 bulk convert, 4 elements per thread (n must be multiple of 4)
// ---------------------------------------------------------------------------
__global__ __launch_bounds__(256) void cvt_bf16_kernel(const float* __restrict__ src,
                                                       __bf16* __restrict__ dst,
                                                       long long n) {
    long long i = ((long long)blockIdx.x * 256 + threadIdx.x) * 4;
    if (i + 3 >= n) return;
    float4 v = ((const float4*)(src + i))[0];
    ushort4 o;
    o.x = __builtin_bit_cast(unsigned short, f2bf(v.x));
    o.y = __builtin_bit_cast(unsigned short, f2bf(v.y));
    o.z = __builtin_bit_cast(unsigned short, f2bf(v.z));
    o.w = __builtin_bit_cast(unsigned short, f2bf(v.w));
    ((ushort4*)(dst + i))[0] = o;
}

// ---------------------------------------------------------------------------
// Row L2 norms: one wave (32 lanes) per row of length D_.
// ---------------------------------------------------------------------------
__global__ __launch_bounds__(256) void rownorm_kernel(const float* __restrict__ x,
                                                      float* __restrict__ nrm,
                                                      float* __restrict__ rnrm,
                                                      int rows) {
    int wid  = (int)((blockIdx.x * blockDim.x + threadIdx.x) >> 5);
    int lane = threadIdx.x & 31;
    if (wid >= rows) return;
    const float* r = x + (size_t)wid * D_;
    float s = 0.f;
    #pragma unroll
    for (int i = 0; i < D_ / 32; ++i) {
        float v = r[lane + i * 32];
        s += v * v;
    }
    #pragma unroll
    for (int off = 16; off > 0; off >>= 1) s += __shfl_xor(s, off, 32);
    if (lane == 0) {
        float n = sqrtf(s);
        nrm[wid] = n;
        if (rnrm) rnrm[wid] = 1.f / fmaxf(n, 1e-20f);
    }
}

__global__ void fillf_kernel(float* __restrict__ p, float v, int n) {
    int i = blockIdx.x * blockDim.x + threadIdx.x;
    if (i < n) p[i] = v;
}
__global__ void filli_kernel(int* __restrict__ p, int v, int n) {
    int i = blockIdx.x * blockDim.x + threadIdx.x;
    if (i < n) p[i] = v;
}

// ---------------------------------------------------------------------------
// Cosine GEMM via v_wmma_f32_16x16x32_bf16 with native bf16 operands.
// out[b, m, n] = dot(proto[b,m,:], fmat[...,n,:]) / max(pnorm[b,m]*fnorm[n], EPS)
// Block: 256 threads = 8 waves; wave w -> 32(M) x 16(N) tile; block tile 32x128.
// Fragment layouts per CDNA5 ISA 7.12.2:
//   A (16x32 bf16): lane l<16 -> row l, K = {0..7, 16..23}; lanes 16-31: K+8
//   B (32x16 bf16): lane l<16 -> col l, K = {0..7, 16..23}; lanes 16-31: col l-16, K+8
//   C (16x16 f32):  vgpr r, lanes 0-15 -> (M=r, N=lane); lanes 16-31 -> (M=8+r)
// ---------------------------------------------------------------------------
__global__ __launch_bounds__(256) void cos_gemm_bf16_kernel(const __bf16* __restrict__ protob,
                                                            const __bf16* __restrict__ fmatb,
                                                            const float* __restrict__ pnorm,
                                                            const float* __restrict__ fnorm,
                                                            float* __restrict__ out,
                                                            int NR, int batchF) {
    int wave = threadIdx.x >> 5;
    int lane = threadIdx.x & 31;
    int b    = blockIdx.y;
    int n0   = blockIdx.x * 128 + wave * 16;

    const __bf16* pB = protob + (size_t)b * P_ * D_;
    const __bf16* fB = fmatb + (batchF ? (size_t)b * NR * D_ : (size_t)0);
    int fnBase       = batchF ? b * NR : 0;

    int lr = lane & 15;
    int kh = lane >> 4;

    const __bf16* arow0 = pB + (size_t)lr * D_;
    const __bf16* arow1 = pB + (size_t)(16 + lr) * D_;
    const __bf16* brow  = fB + (size_t)(n0 + lr) * D_;

    v8f acc0 = {0.f, 0.f, 0.f, 0.f, 0.f, 0.f, 0.f, 0.f};
    v8f acc1 = {0.f, 0.f, 0.f, 0.f, 0.f, 0.f, 0.f, 0.f};

    int ka = kh * 8;
    int kb = kh * 8;

    #pragma unroll 2
    for (int k0 = 0; k0 < D_; k0 += 32) {
        v8bf a0lo = *(const v8bf*)(arow0 + k0 + ka);
        v8bf a0hi = *(const v8bf*)(arow0 + k0 + 16 + ka);
        v8bf a1lo = *(const v8bf*)(arow1 + k0 + ka);
        v8bf a1hi = *(const v8bf*)(arow1 + k0 + 16 + ka);
        v8bf blo  = *(const v8bf*)(brow + k0 + kb);
        v8bf bhi  = *(const v8bf*)(brow + k0 + 16 + kb);
        v16bf bb  = __builtin_shufflevector(blo, bhi, 0, 1, 2, 3, 4, 5, 6, 7,
                                            8, 9, 10, 11, 12, 13, 14, 15);
        __builtin_prefetch(brow + k0 + 512, 0, 0);
        v16bf a0 = __builtin_shufflevector(a0lo, a0hi, 0, 1, 2, 3, 4, 5, 6, 7,
                                           8, 9, 10, 11, 12, 13, 14, 15);
        v16bf a1 = __builtin_shufflevector(a1lo, a1hi, 0, 1, 2, 3, 4, 5, 6, 7,
                                           8, 9, 10, 11, 12, 13, 14, 15);
        acc0 = __builtin_amdgcn_wmma_f32_16x16x32_bf16(false, a0, false, bb,
                                                       (short)0, acc0, false, false);
        acc1 = __builtin_amdgcn_wmma_f32_16x16x32_bf16(false, a1, false, bb,
                                                       (short)0, acc1, false, false);
    }

    int nc   = n0 + lr;
    float fn = fnorm[fnBase + nc];
    int mh   = kh * 8;
    #pragma unroll
    for (int r = 0; r < 8; ++r) {
        int m0 = mh + r;
        int m1 = m0 + 16;
        float d0 = fmaxf(pnorm[b * P_ + m0] * fn, EPS_);
        float d1 = fmaxf(pnorm[b * P_ + m1] * fn, EPS_);
        out[((size_t)(b * P_ + m0)) * NR + nc] = acc0[r] / d0;
        out[((size_t)(b * P_ + m1)) * NR + nc] = acc1[r] / d1;
    }
}

// Fallback GEMM: fp32 operands converted in-flight (used only if the workspace
// cannot hold the bf16 mirror of feats).
static __device__ __forceinline__ void load8bf(v16bf& f, int base, const float* __restrict__ p) {
    float4 a = ((const float4*)p)[0];
    float4 b = ((const float4*)p)[1];
    f[base + 0] = f2bf(a.x); f[base + 1] = f2bf(a.y);
    f[base + 2] = f2bf(a.z); f[base + 3] = f2bf(a.w);
    f[base + 4] = f2bf(b.x); f[base + 5] = f2bf(b.y);
    f[base + 6] = f2bf(b.z); f[base + 7] = f2bf(b.w);
}

__global__ __launch_bounds__(256) void cos_gemm_f32_kernel(const float* __restrict__ proto,
                                                           const float* __restrict__ fmat,
                                                           const float* __restrict__ pnorm,
                                                           const float* __restrict__ fnorm,
                                                           float* __restrict__ out,
                                                           int NR, int batchF) {
    int wave = threadIdx.x >> 5;
    int lane = threadIdx.x & 31;
    int b    = blockIdx.y;
    int n0   = blockIdx.x * 128 + wave * 16;

    const float* pB = proto + (size_t)b * P_ * D_;
    const float* fB = fmat + (batchF ? (size_t)b * NR * D_ : (size_t)0);
    int fnBase      = batchF ? b * NR : 0;

    int lr = lane & 15;
    int kh = lane >> 4;

    const float* arow0 = pB + (size_t)lr * D_;
    const float* arow1 = pB + (size_t)(16 + lr) * D_;
    const float* brow  = fB + (size_t)(n0 + lr) * D_;

    v8f acc0 = {0.f, 0.f, 0.f, 0.f, 0.f, 0.f, 0.f, 0.f};
    v8f acc1 = {0.f, 0.f, 0.f, 0.f, 0.f, 0.f, 0.f, 0.f};

    int ka = kh * 8;
    int kb = kh * 8;

    for (int k0 = 0; k0 < D_; k0 += 32) {
        v16bf a0, a1, bb;
        load8bf(a0, 0, arow0 + k0 + ka);
        load8bf(a0, 8, arow0 + k0 + 16 + ka);
        load8bf(a1, 0, arow1 + k0 + ka);
        load8bf(a1, 8, arow1 + k0 + 16 + ka);
        load8bf(bb, 0, brow + k0 + kb);
        load8bf(bb, 8, brow + k0 + 16 + kb);
        acc0 = __builtin_amdgcn_wmma_f32_16x16x32_bf16(false, a0, false, bb,
                                                       (short)0, acc0, false, false);
        acc1 = __builtin_amdgcn_wmma_f32_16x16x32_bf16(false, a1, false, bb,
                                                       (short)0, acc1, false, false);
    }

    int nc   = n0 + lr;
    float fn = fnorm[fnBase + nc];
    int mh   = kh * 8;
    #pragma unroll
    for (int r = 0; r < 8; ++r) {
        int m0 = mh + r;
        int m1 = m0 + 16;
        float d0 = fmaxf(pnorm[b * P_ + m0] * fn, EPS_);
        float d1 = fmaxf(pnorm[b * P_ + m1] * fn, EPS_);
        out[((size_t)(b * P_ + m0)) * NR + nc] = acc0[r] / d0;
        out[((size_t)(b * P_ + m1)) * NR + nc] = acc1[r] / d1;
    }
}

// ---------------------------------------------------------------------------
// Per (b,p) softmax stats over N: row max and reciprocal of sum(exp(s*(x-max))).
// ---------------------------------------------------------------------------
__global__ __launch_bounds__(256) void rowstats_kernel(const float* __restrict__ sim,
                                                       const float* __restrict__ sscale,
                                                       float* __restrict__ mxv,
                                                       float* __restrict__ invZ) {
    int row = blockIdx.x;  // b*P + p
    const float* r = sim + (size_t)row * N_;
    float s = sscale[row];
    __shared__ float red[256];

    float m = -1e30f;
    for (int i = threadIdx.x; i < N_; i += 256) m = fmaxf(m, r[i]);
    red[threadIdx.x] = m;
    __syncthreads();
    for (int off = 128; off > 0; off >>= 1) {
        if ((int)threadIdx.x < off)
            red[threadIdx.x] = fmaxf(red[threadIdx.x], red[threadIdx.x + off]);
        __syncthreads();
    }
    m = red[0];
    __syncthreads();

    float sum = 0.f;
    for (int i = threadIdx.x; i < N_; i += 256) sum += __expf(s * (r[i] - m));
    red[threadIdx.x] = sum;
    __syncthreads();
    for (int off = 128; off > 0; off >>= 1) {
        if ((int)threadIdx.x < off)
            red[threadIdx.x] += red[threadIdx.x + off];
        __syncthreads();
    }
    if (threadIdx.x == 0) {
        mxv[row]  = m;
        invZ[row] = 1.f / red[0];
    }
}

// ---------------------------------------------------------------------------
// Per (b,n): argmax over P of softmax weight; record index + winning weight,
// histogram counts per prototype.
// ---------------------------------------------------------------------------
__global__ __launch_bounds__(256) void argmax_kernel(const float* __restrict__ sim,
                                                     const float* __restrict__ mxv,
                                                     const float* __restrict__ invZ,
                                                     const float* __restrict__ sscale,
                                                     int* __restrict__ idx,
                                                     float* __restrict__ wval,
                                                     int* __restrict__ count) {
    int b = blockIdx.y;
    int n = blockIdx.x * 256 + threadIdx.x;
    __shared__ float smx[P_], siz[P_], ssc[P_];
    if (threadIdx.x < P_) {
        smx[threadIdx.x] = mxv[b * P_ + threadIdx.x];
        siz[threadIdx.x] = invZ[b * P_ + threadIdx.x];
        ssc[threadIdx.x] = sscale[b * P_ + threadIdx.x];
    }
    __syncthreads();

    const float* base = sim + (size_t)b * P_ * N_ + n;
    float best = -1.f;
    int bp = 0;
    #pragma unroll 4
    for (int p = 0; p < P_; ++p) {
        float w = __expf(ssc[p] * (base[(size_t)p * N_] - smx[p])) * siz[p];
        if (w > best) { best = w; bp = p; }   // strict > keeps first max (argmax semantics)
    }
    idx[b * N_ + n]  = bp;
    wval[b * N_ + n] = best;
    atomicAdd(&count[b * P_ + bp], 1);
}

// ---------------------------------------------------------------------------
// Prototype update + unit-vector sums. Chunk of 64 d-columns per block,
// 4 independent n-streams with private LDS copies (deterministic, atomic-free:
// each thread owns one (stream, d) column). Grid: (D/64, B) = 384 blocks.
//   proto_new[b,p,:]  = sum_{n: idx==p} wval[n] * feats[b,n,:]
//   sumunit[b,p,:]    = sum_{n: idx==p} feats[b,n,:] / ||feats[b,n]||
// ---------------------------------------------------------------------------
#define CHK_ 64
__global__ __launch_bounds__(256) void accum_kernel(const float* __restrict__ feats,
                                                    const int* __restrict__ idx,
                                                    const float* __restrict__ wval,
                                                    const float* __restrict__ rnf,
                                                    float* __restrict__ proto,
                                                    float* __restrict__ sumunit) {
    __shared__ float accW[4][P_][CHK_];
    __shared__ float accU[4][P_][CHK_];
    int b  = blockIdx.y;
    int cb = blockIdx.x * CHK_;
    int d  = threadIdx.x & (CHK_ - 1);
    int st = threadIdx.x >> 6;  // n-stream 0..3

    for (int i = threadIdx.x; i < 4 * P_ * CHK_; i += 256) {
        (&accW[0][0][0])[i] = 0.f;
        (&accU[0][0][0])[i] = 0.f;
    }
    __syncthreads();

    const float* fb = feats + (size_t)b * N_ * D_ + cb;
    const int*   ib = idx + b * N_;
    const float* wb = wval + b * N_;
    const float* rb = rnf + b * N_;
    for (int n = st; n < N_; n += 4) {
        int   p  = ib[n];
        float w  = wb[n];
        float ru = rb[n];
        float f  = fb[(size_t)n * D_ + d];
        accW[st][p][d] += w * f;   // unique (st,p-slot,d) per thread: race-free
        accU[st][p][d] += ru * f;
    }
    __syncthreads();

    for (int i = threadIdx.x; i < P_ * CHK_; i += 256) {
        int p  = i / CHK_;
        int dd = i % CHK_;
        float w = accW[0][p][dd] + accW[1][p][dd] + accW[2][p][dd] + accW[3][p][dd];
        float u = accU[0][p][dd] + accU[1][p][dd] + accU[2][p][dd] + accU[3][p][dd];
        proto[((size_t)(b * P_ + p)) * D_ + cb + dd]   = w;
        sumunit[((size_t)(b * P_ + p)) * D_ + cb + dd] = u;
    }
}

// ---------------------------------------------------------------------------
// Per (b,p): new norm, density, tau -> softmax scale for next iteration.
// ---------------------------------------------------------------------------
__global__ __launch_bounds__(256) void tau_kernel(const float* __restrict__ proto,
                                                  const float* __restrict__ sumunit,
                                                  const int* __restrict__ count,
                                                  float* __restrict__ pnorm,
                                                  float* __restrict__ sscale) {
    int row = blockIdx.x;  // b*P + p
    const float* pr = proto + (size_t)row * D_;
    const float* su = sumunit + (size_t)row * D_;
    __shared__ float r1[256], r2[256];
    float a = 0.f, c = 0.f;
    for (int i = threadIdx.x; i < D_; i += 256) {
        float v = pr[i];
        a += v * v;
        c += v * su[i];
    }
    r1[threadIdx.x] = a;
    r2[threadIdx.x] = c;
    __syncthreads();
    for (int off = 128; off > 0; off >>= 1) {
        if ((int)threadIdx.x < off) {
            r1[threadIdx.x] += r1[threadIdx.x + off];
            r2[threadIdx.x] += r2[threadIdx.x + off];
        }
        __syncthreads();
    }
    if (threadIdx.x == 0) {
        float pn = sqrtf(r1[0]);
        pnorm[row] = pn;
        float dens = r2[0] / fmaxf(pn, EPS_);
        float ms = (float)count[row];
        float density = 1.f - ((ms >= 1.f) ? dens / ms : 0.f);
        float tau = fmaxf(density, 1e-10f);
        sscale[row] = 1.f / (TEMP_ * tau);
    }
}

// ---------------------------------------------------------------------------
extern "C" void kernel_launch(void* const* d_in, const int* in_sizes, int n_in,
                              void* d_out, int out_size, void* d_ws, size_t ws_size,
                              hipStream_t stream) {
    (void)in_sizes; (void)n_in; (void)out_size;
    const float* prototypes = (const float*)d_in[0];  // [B,P,D]
    const float* feats      = (const float*)d_in[1];  // [B,N,D]
    const float* feats_org  = (const float*)d_in[2];  // [M,D]
    float* out = (float*)d_out;                       // [B*P*D] ++ [B*P*M]

    char* ws = (char*)d_ws;
    size_t off = 0;
    auto alloc = [&](size_t bytes) -> void* {
        void* p = ws + off;
        off = (off + bytes + 255) & ~(size_t)255;
        return p;
    };
    // Small buffers first; the big bf16 feats mirror last (optional).
    float*  proto   = (float*)alloc((size_t)B_ * P_ * D_ * 4);
    float*  sumunit = (float*)alloc((size_t)B_ * P_ * D_ * 4);
    float*  sim     = (float*)alloc((size_t)B_ * P_ * N_ * 4);
    float*  nf      = (float*)alloc((size_t)B_ * N_ * 4);
    float*  rnf     = (float*)alloc((size_t)B_ * N_ * 4);
    float*  norg    = (float*)alloc((size_t)M_ * 4);
    float*  pnorm   = (float*)alloc((size_t)B_ * P_ * 4);
    float*  sscale  = (float*)alloc((size_t)B_ * P_ * 4);
    float*  mxv     = (float*)alloc((size_t)B_ * P_ * 4);
    float*  invZ    = (float*)alloc((size_t)B_ * P_ * 4);
    int*    idx     = (int*)alloc((size_t)B_ * N_ * 4);
    float*  wval    = (float*)alloc((size_t)B_ * N_ * 4);
    int*    count   = (int*)alloc((size_t)B_ * P_ * 4);
    __bf16* protob  = (__bf16*)alloc((size_t)B_ * P_ * D_ * 2);
    __bf16* forgb   = (__bf16*)alloc((size_t)M_ * D_ * 2);
    __bf16* featsb  = (__bf16*)alloc((size_t)B_ * N_ * D_ * 2);
    bool bigws = (off <= ws_size);  // can we keep the bf16 feats mirror?

    const long long nProto = (long long)B_ * P_ * D_;
    const long long nFeats = (long long)B_ * N_ * D_;
    const long long nForg  = (long long)M_ * D_;

    // proto working copy (input must not be mutated)
    hipMemcpyAsync(proto, prototypes, (size_t)nProto * 4, hipMemcpyDeviceToDevice, stream);

    // bf16 mirrors
    cvt_bf16_kernel<<<(unsigned)((nForg / 4 + 255) / 256), 256, 0, stream>>>(feats_org, forgb, nForg);
    if (bigws)
        cvt_bf16_kernel<<<(unsigned)((nFeats / 4 + 255) / 256), 256, 0, stream>>>(feats, featsb, nFeats);
    cvt_bf16_kernel<<<(unsigned)((nProto / 4 + 255) / 256), 256, 0, stream>>>(proto, protob, nProto);

    // Norms of feats rows (+ reciprocals), feats_org rows, initial prototypes
    rownorm_kernel<<<(B_ * N_) / 8, 256, 0, stream>>>(feats, nf, rnf, B_ * N_);
    rownorm_kernel<<<M_ / 8, 256, 0, stream>>>(feats_org, norg, nullptr, M_);
    rownorm_kernel<<<(B_ * P_) / 8, 256, 0, stream>>>(proto, pnorm, nullptr, B_ * P_);
    // initial temperature scale: 1/(TEMP*TAU0) = 1/(0.1*0.1) = 100
    fillf_kernel<<<(B_ * P_ + 255) / 256, 256, 0, stream>>>(sscale, 100.f, B_ * P_);

    for (int it = 0; it < 5; ++it) {
        if (bigws)
            cos_gemm_bf16_kernel<<<dim3(N_ / 128, B_), 256, 0, stream>>>(
                protob, featsb, pnorm, nf, sim, N_, 1);
        else
            cos_gemm_f32_kernel<<<dim3(N_ / 128, B_), 256, 0, stream>>>(
                proto, feats, pnorm, nf, sim, N_, 1);
        rowstats_kernel<<<B_ * P_, 256, 0, stream>>>(sim, sscale, mxv, invZ);
        filli_kernel<<<(B_ * P_ + 255) / 256, 256, 0, stream>>>(count, 0, B_ * P_);
        argmax_kernel<<<dim3(N_ / 256, B_), 256, 0, stream>>>(
            sim, mxv, invZ, sscale, idx, wval, count);
        accum_kernel<<<dim3(D_ / CHK_, B_), 256, 0, stream>>>(
            feats, idx, wval, rnf, proto, sumunit);
        cvt_bf16_kernel<<<(unsigned)((nProto / 4 + 255) / 256), 256, 0, stream>>>(proto, protob, nProto);
        tau_kernel<<<B_ * P_, 256, 0, stream>>>(proto, sumunit, count, pnorm, sscale);
    }

    // Final cosine similarity vs feats_org -> out[B*P*D ...]
    cos_gemm_bf16_kernel<<<dim3(M_ / 128, B_), 256, 0, stream>>>(
        protob, forgb, pnorm, norg, out + (size_t)B_ * P_ * D_, M_, 0);
    // Final prototypes -> out[0 .. B*P*D)
    hipMemcpyAsync(out, proto, (size_t)nProto * 4, hipMemcpyDeviceToDevice, stream);
}
